// SparseMoE_40364102648332
// MI455X (gfx1250) — matrix operand order repaired and checked by reading, not compiled, not run on hardware
//
#include <hip/hip_runtime.h>
#include <hip/hip_bf16.h>
#include <math.h>

// ---- problem constants (from reference) ----
#define NTOK   8192      // B*H*W
#define DIM    384
#define NEXP   8
#define HIDN   1536
#define ODIM   384
// ---- tiling ----
#define BM     64        // token rows per block (4 row-tiles)
#define KC     64        // hidden-chunk width (4 col-tiles)
#define XPITCH 400       // f16 elems per Xs row (mult of 8 -> 16B-aligned b128 lds loads)
#define HPITCH 72        // f16 elems per Hs row

typedef _Float16 h16 __attribute__((ext_vector_type(16)));
typedef _Float16 h8v __attribute__((ext_vector_type(8)));
typedef float    f8  __attribute__((ext_vector_type(8)));

// ---- workspace layout (bytes) ----
#define WS_COUNTS 0
#define WS_TOK    256
#define WS_GATE   (WS_TOK + NEXP * NTOK * 4)
#define WS_W1H    ((WS_GATE + NEXP * NTOK * 4 + 255) & ~255)
#define W1_ELEMS  ((size_t)NEXP * DIM * HIDN)
#define W2_ELEMS  ((size_t)NEXP * HIDN * ODIM)
#define WS_W2H    (WS_W1H + W1_ELEMS * 2)
#define WS_NEED   (WS_W2H + W2_ELEMS * 2)

// ---------------- kernel 0: zero output + expert counters ----------------
__global__ void moe_zero(float* __restrict__ out, int n, int* __restrict__ counts) {
    int i = blockIdx.x * blockDim.x + threadIdx.x;
    if (i < n) out[i] = 0.0f;
    if (i < NEXP) counts[i] = 0;
}

// ---------------- kernel 0b: one-time f32 -> f16 weight conversion ----------------
__global__ void cvt_f32_f16(const float* __restrict__ in, _Float16* __restrict__ out, int n) {
    int i = (blockIdx.x * blockDim.x + threadIdx.x) * 8;
    if (i >= n) return;
    float4 a = ((const float4*)(in + i))[0];
    float4 b = ((const float4*)(in + i))[1];
    h8v v;
    v[0] = (_Float16)a.x; v[1] = (_Float16)a.y; v[2] = (_Float16)a.z; v[3] = (_Float16)a.w;
    v[4] = (_Float16)b.x; v[5] = (_Float16)b.y; v[6] = (_Float16)b.z; v[7] = (_Float16)b.w;
    *(h8v*)(out + i) = v;
}

// ---------------- kernel 1: router (one wave32 per token) ----------------
__global__ void moe_router(const float* __restrict__ x,
                           const float* __restrict__ Wg,
                           const float* __restrict__ bg,
                           int* __restrict__ counts,
                           int* __restrict__ tokList,
                           float* __restrict__ gateList) {
    const int wave = threadIdx.x >> 5;
    const int lane = threadIdx.x & 31;
    const int t = blockIdx.x * 8 + wave;
    if (t >= NTOK) return;

    float acc[NEXP];
#pragma unroll
    for (int e = 0; e < NEXP; ++e) acc[e] = 0.0f;

    const float* xr = x + (size_t)t * DIM;
    for (int d = lane; d < DIM; d += 32) {
        float xv = xr[d];
        const float4* wr = (const float4*)(Wg + (size_t)d * NEXP);
        float4 w0 = wr[0], w1 = wr[1];
        acc[0] += xv * w0.x; acc[1] += xv * w0.y;
        acc[2] += xv * w0.z; acc[3] += xv * w0.w;
        acc[4] += xv * w1.x; acc[5] += xv * w1.y;
        acc[6] += xv * w1.z; acc[7] += xv * w1.w;
    }
#pragma unroll
    for (int m = 16; m >= 1; m >>= 1) {
#pragma unroll
        for (int e = 0; e < NEXP; ++e) acc[e] += __shfl_xor(acc[e], m, 32);
    }

    if (lane == 0) {
        float l[NEXP];
#pragma unroll
        for (int e = 0; e < NEXP; ++e) l[e] = acc[e] + bg[e];
        int i1 = 0; float v1 = l[0];
#pragma unroll
        for (int e = 1; e < NEXP; ++e) if (l[e] > v1) { v1 = l[e]; i1 = e; }
        int i2 = -1; float v2 = -3.0e38f;
#pragma unroll
        for (int e = 0; e < NEXP; ++e) if (e != i1 && l[e] > v2) { v2 = l[e]; i2 = e; }
        float e2 = expf(v2 - v1);
        float inv = 1.0f / (1.0f + e2);
        float g1 = inv, g2 = e2 * inv;

        int p1 = atomicAdd(&counts[i1], 1);
        tokList[i1 * NTOK + p1] = t;  gateList[i1 * NTOK + p1] = g1;
        int p2 = atomicAdd(&counts[i2], 1);
        tokList[i2 * NTOK + p2] = t;  gateList[i2 * NTOK + p2] = g2;
    }
}

// ---- A-fragment loader: 16-bit A VGPR layout from row-major f16 LDS tile ----
__device__ __forceinline__ h16 load_a_frag(const _Float16* lds, int pitch,
                                           int row0, int k, int lane) {
    const _Float16* p = lds + (size_t)(row0 + (lane & 15)) * pitch + k + ((lane >> 4) << 3);
    h8v r0 = *(const h8v*)(p);
    h8v r1 = *(const h8v*)(p + 16);
    return __builtin_shufflevector(r0, r1, 0, 1, 2, 3, 4, 5, 6, 7,
                                           8, 9, 10, 11, 12, 13, 14, 15);
}

// ---- B-fragment loaders: lane = N col, elems along K (rows 8h+s, 16+8h+s) ----
__device__ __forceinline__ h16 load_b_f16(const _Float16* p, int pitch) {
    h16 b;
#pragma unroll
    for (int s = 0; s < 8; ++s) { b[s] = p[(size_t)s * pitch]; b[8 + s] = p[(size_t)(16 + s) * pitch]; }
    return b;
}
__device__ __forceinline__ h16 load_b_f32cvt(const float* p, int pitch) {
    h16 b;
#pragma unroll
    for (int s = 0; s < 8; ++s) { b[s] = (_Float16)p[(size_t)s * pitch]; b[8 + s] = (_Float16)p[(size_t)(16 + s) * pitch]; }
    return b;
}

__device__ __forceinline__ float gelu_exact(float v) {
    return 0.5f * v * (1.0f + erff(v * 0.70710678118654752f));
}

#define WMMA_F16(a, b, c) \
    __builtin_amdgcn_wmma_f32_16x16x32_f16(false, (a), false, (b), (short)0, (c), false, false)

// ---------------- kernel 2: fused expert FFN ----------------
// gather(64 tok) -> [X@W1 -> +b1 -> GELU -> @W2] over 24 KC-chunks -> +b2, *gate, scatter
template <bool F16W>
__global__ __launch_bounds__(256, 1) void moe_expert(
        const float* __restrict__ x,
        const void* __restrict__ W1v, const float* __restrict__ b1,
        const void* __restrict__ W2v, const float* __restrict__ b2,
        const int* __restrict__ counts,
        const int* __restrict__ tokList, const float* __restrict__ gateList,
        float* __restrict__ out) {

    __shared__ _Float16 Xs[BM * XPITCH];   // 51200 B
    __shared__ _Float16 Hs[BM * HPITCH];   //  9216 B
    __shared__ int   tokS[BM];
    __shared__ float gateS[BM];

    const int e = blockIdx.y;
    const int cnt = counts[e];
    const int mstart = blockIdx.x * BM;
    if (mstart >= cnt) return;            // block-uniform exit; EXEC stays full after

    const int tid  = threadIdx.x;
    const int wave = tid >> 5;
    const int lane = tid & 31;

    if (tid < BM) {
        int idx = mstart + tid;
        if (idx < cnt) { tokS[tid] = tokList[e * NTOK + idx]; gateS[tid] = gateList[e * NTOK + idx]; }
        else           { tokS[tid] = 0;                       gateS[tid] = 0.0f; }
    }
    __syncthreads();

    // ---- gather X tile -> LDS as f16 (64 rows x 384, 4 threads per row) ----
    {
        int row = tid >> 2;
        int seg = (tid & 3) * 96;
        const float4* src = (const float4*)(x + (size_t)tokS[row] * DIM + seg);
        _Float16* dst = Xs + (size_t)row * XPITCH + seg;
#pragma unroll
        for (int i = 0; i < 12; ++i) {
            float4 a = src[2 * i], b = src[2 * i + 1];
            h8v v;
            v[0] = (_Float16)a.x; v[1] = (_Float16)a.y; v[2] = (_Float16)a.z; v[3] = (_Float16)a.w;
            v[4] = (_Float16)b.x; v[5] = (_Float16)b.y; v[6] = (_Float16)b.z; v[7] = (_Float16)b.w;
            *(h8v*)(dst + 8 * i) = v;
        }
    }
    __syncthreads();

    // wave -> tile mapping: pair p = rows [32p, 32p+32); cq = col quarter
    const int p  = wave >> 2;         // 0..1 : row-tile pair {2p, 2p+1}
    const int cq = wave & 3;          // 0..3 : stage1 H col-tile / stage2 col group

    const _Float16* W1h = (const _Float16*)W1v + (F16W ? (size_t)e * DIM * HIDN : 0);
    const _Float16* W2h = (const _Float16*)W2v + (F16W ? (size_t)e * HIDN * ODIM : 0);
    const float*    W1f = (const float*)W1v + (size_t)e * DIM * HIDN;
    const float*    W2f = (const float*)W2v + (size_t)e * HIDN * ODIM;

    f8 acc[2][6];
#pragma unroll
    for (int rp = 0; rp < 2; ++rp)
#pragma unroll
        for (int j = 0; j < 6; ++j) acc[rp][j] = (f8){};

    const int n15 = lane & 15;
    const int mhi = (lane >> 4) * 8;
    const int r0  = (2 * p) * 16;     // first row-tile base
    const int r1  = r0 + 16;          // second row-tile base

    for (int c = 0; c < HIDN / KC; ++c) {
        const int hcol = c * KC + cq * 16;

        auto loadB1 = [&](int k) -> h16 {
            if constexpr (F16W) return load_b_f16(W1h + (size_t)(k + ((lane >> 4) << 3)) * HIDN + hcol + n15, HIDN);
            else                return load_b_f32cvt(W1f + (size_t)(k + ((lane >> 4) << 3)) * HIDN + hcol + n15, HIDN);
        };

        // ---- stage 1: H(64xKC) = gelu(X @ W1[:, chunk] + b1); this wave: 2 tiles ----
        // groups of 3 k-steps: 3 B-frags issued together, first uses 1..3, last uses 4..6
        f8 h0 = (f8){}, h1 = (f8){};
#pragma unroll
        for (int kg = 0; kg < DIM; kg += 96) {
            h16 b0 = loadB1(kg);
            h16 b1f = loadB1(kg + 32);
            h16 b2f = loadB1(kg + 64);
            h16 a00 = load_a_frag(Xs, XPITCH, r0, kg,      lane);
            h16 a10 = load_a_frag(Xs, XPITCH, r0, kg + 32, lane);
            h16 a20 = load_a_frag(Xs, XPITCH, r0, kg + 64, lane);
            h0 = WMMA_F16(a00, b0,  h0);
            h0 = WMMA_F16(a10, b1f, h0);
            h0 = WMMA_F16(a20, b2f, h0);
            h16 a01 = load_a_frag(Xs, XPITCH, r1, kg,      lane);
            h16 a11 = load_a_frag(Xs, XPITCH, r1, kg + 32, lane);
            h16 a21 = load_a_frag(Xs, XPITCH, r1, kg + 64, lane);
            h1 = WMMA_F16(a01, b0,  h1);
            h1 = WMMA_F16(a11, b1f, h1);
            h1 = WMMA_F16(a21, b2f, h1);
        }
        float b1v = b1[(size_t)e * HIDN + hcol + n15];
#pragma unroll
        for (int r = 0; r < 8; ++r) {
            float v0 = gelu_exact(h0[r] + b1v);
            float v1 = gelu_exact(h1[r] + b1v);
            Hs[(size_t)(r0 + mhi + r) * HPITCH + cq * 16 + n15] = (_Float16)v0;
            Hs[(size_t)(r1 + mhi + r) * HPITCH + cq * 16 + n15] = (_Float16)v1;
        }
        __syncthreads();

        // ---- stage 2: O(64x384) += H(64xKC) @ W2[chunk, :]; this wave: 12 tiles ----
        h16 a00 = load_a_frag(Hs, HPITCH, r0, 0,  lane);
        h16 a01 = load_a_frag(Hs, HPITCH, r1, 0,  lane);
        h16 a10 = load_a_frag(Hs, HPITCH, r0, 32, lane);
        h16 a11 = load_a_frag(Hs, HPITCH, r1, 32, lane);
        const size_t w2r0 = (size_t)(c * KC + ((lane >> 4) << 3)) * ODIM;
        const size_t w2r1 = (size_t)(c * KC + 32 + ((lane >> 4) << 3)) * ODIM;

        auto loadB2 = [&](size_t off) -> h16 {
            if constexpr (F16W) return load_b_f16(W2h + off + n15, ODIM);
            else                return load_b_f32cvt(W2f + off + n15, ODIM);
        };

        // groups of 2 output columns x 2 k-steps: 4 B-frags issued together;
        // first uses at WMMA 1..4, last uses at 5..8 -> all four live at once.
#pragma unroll
        for (int jj = 0; jj < 6; jj += 2) {
            size_t colA = (size_t)((cq * 6 + jj) * 16);
            size_t colB = colA + 16;
            h16 b00 = loadB2(w2r0 + colA);   // j=jj,   k=0..31
            h16 b01 = loadB2(w2r1 + colA);   // j=jj,   k=32..63
            h16 b10 = loadB2(w2r0 + colB);   // j=jj+1, k=0..31
            h16 b11 = loadB2(w2r1 + colB);   // j=jj+1, k=32..63
            acc[0][jj]     = WMMA_F16(a00, b00, acc[0][jj]);
            acc[0][jj]     = WMMA_F16(a10, b01, acc[0][jj]);
            acc[0][jj + 1] = WMMA_F16(a00, b10, acc[0][jj + 1]);
            acc[0][jj + 1] = WMMA_F16(a10, b11, acc[0][jj + 1]);
            acc[1][jj]     = WMMA_F16(a01, b00, acc[1][jj]);
            acc[1][jj]     = WMMA_F16(a11, b01, acc[1][jj]);
            acc[1][jj + 1] = WMMA_F16(a01, b10, acc[1][jj + 1]);
            acc[1][jj + 1] = WMMA_F16(a11, b11, acc[1][jj + 1]);
        }
        __syncthreads();   // protect Hs before next chunk's stage-1 writes
    }

    // ---- epilogue: add b2, scale by gate, scatter-add ----
#pragma unroll
    for (int rp = 0; rp < 2; ++rp) {
#pragma unroll
        for (int j = 0; j < 6; ++j) {
            int col = (cq * 6 + j) * 16 + n15;
            float b2v = b2[(size_t)e * ODIM + col];
#pragma unroll
            for (int r = 0; r < 8; ++r) {
                int m = (2 * p + rp) * 16 + mhi + r;
                float g = gateS[m];
                atomicAdd(out + (size_t)tokS[m] * ODIM + col, (acc[rp][j][r] + b2v) * g);
            }
        }
    }
}

// ---------------- host entry ----------------
extern "C" void kernel_launch(void* const* d_in, const int* in_sizes, int n_in,
                              void* d_out, int out_size, void* d_ws, size_t ws_size,
                              hipStream_t stream) {
    const float* x  = (const float*)d_in[0];
    const float* Wg = (const float*)d_in[1];
    const float* bg = (const float*)d_in[2];
    const float* W1 = (const float*)d_in[3];
    const float* b1 = (const float*)d_in[4];
    const float* W2 = (const float*)d_in[5];
    const float* b2 = (const float*)d_in[6];
    float* out = (float*)d_out;

    char* ws = (char*)d_ws;
    int*      counts   = (int*)(ws + WS_COUNTS);
    int*      tokList  = (int*)(ws + WS_TOK);
    float*    gateList = (float*)(ws + WS_GATE);
    _Float16* W1h      = (_Float16*)(ws + WS_W1H);
    _Float16* W2h      = (_Float16*)(ws + WS_W2H);

    // 1) zero output + counters
    int nOut = NTOK * ODIM;
    moe_zero<<<(nOut + 255) / 256, 256, 0, stream>>>(out, nOut, counts);

    // 2) router
    moe_router<<<NTOK / 8, 256, 0, stream>>>(x, Wg, bg, counts, tokList, gateList);

    // 3) expert FFN — f16-weight fast path if workspace permits
    dim3 grid(NTOK / BM, NEXP);
    if (ws_size >= (size_t)WS_NEED) {
        cvt_f32_f16<<<(int)(W1_ELEMS / 8 + 255) / 256, 256, 0, stream>>>(W1, W1h, (int)W1_ELEMS);
        cvt_f32_f16<<<(int)(W2_ELEMS / 8 + 255) / 256, 256, 0, stream>>>(W2, W2h, (int)W2_ELEMS);
        moe_expert<true><<<grid, 256, 0, stream>>>(x, W1h, b1, W2h, b2,
                                                   counts, tokList, gateList, out);
    } else {
        moe_expert<false><<<grid, 256, 0, stream>>>(x, W1, b1, W2, b2,
                                                    counts, tokList, gateList, out);
    }
}
